// TransformerBlock_63900523430651
// MI455X (gfx1250) — matrix operand order repaired and checked by reading, not compiled, not run on hardware
//
#include <hip/hip_runtime.h>

// ---------------------------------------------------------------------------
// CDNA5 (gfx1250) transformer block.
// - all matmuls: v_wmma_f32_16x16x32_bf16, 16x64 wave tiles (4 accumulators)
// - B tiles staged to LDS by the Tensor Data Mover (tensor_load_to_lds) with
//   descriptor-driven padding for conflict-free ds_load_b128 fragment reads,
//   double-buffered and synchronized with s_wait_tensorcnt.  Guarded fallback
//   uses plain global_load + ds_store staging.
// ---------------------------------------------------------------------------

typedef __bf16 bf16_t;
typedef __attribute__((ext_vector_type(16))) __bf16 v16bf;
typedef __attribute__((ext_vector_type(8)))  __bf16 v8bf;
typedef __attribute__((ext_vector_type(4)))  __bf16 v4bf;
typedef __attribute__((ext_vector_type(8)))  float  v8f;
typedef unsigned int u32x4 __attribute__((ext_vector_type(4)));
typedef int          i32x8 __attribute__((ext_vector_type(8)));
typedef int          i32x4 __attribute__((ext_vector_type(4)));

#define BB   8
#define TT   1024
#define DD   1024
#define HH   16
#define HDM  64
#define DFF  4096
#define MROWS (BB * TT)   // 8192

#if __has_builtin(__builtin_amdgcn_tensor_load_to_lds) && \
    __has_builtin(__builtin_amdgcn_s_wait_tensorcnt)
#define USE_TDM 1
#else
#define USE_TDM 0
#endif

union ABfrag { v16bf v; v8bf h[2]; };

__device__ __forceinline__ v8f wmma_bf16(v16bf a, v16bf b, v8f c) {
  // (neg_a, A, neg_b, B, c_mod, C, reuse_a, reuse_b)
  return __builtin_amdgcn_wmma_f32_16x16x32_bf16(false, a, false, b, (short)0, c,
                                                 false, false);
}

#if USE_TDM
// Issue one TDM 2-D tile load: 64 rows x 32 bf16 (64B/row) from a [N,K] bf16
// matrix (row stride K elems) into LDS, with +16B pad after each 16 DWORDs so
// the LDS image has an 80B row pitch (bank-conflict-free for the B fragments).
__device__ __forceinline__ void tdm_load_tile(unsigned lds_byte,
                                              unsigned long long gaddr,
                                              int k_elems) {
  u32x4 g0;
  g0[0] = 1u;                                   // count=1 (valid descriptor)
  g0[1] = lds_byte;                             // lds_addr (bytes)
  g0[2] = (unsigned)gaddr;                      // global_addr[31:0]
  g0[3] = (unsigned)(gaddr >> 32) | 0x80000000u;// global_addr[56:32] | type=2
  i32x8 g1;
  g1[0] = (1 << 16)       // data_size = 2 bytes
        | (1 << 20)       // pad_enable
        | (3 << 22)       // pad_interval: 16 DWORDs
        | (3 << 25);      // pad_amount:   4 DWORDs (16B)
  const int big = 1 << 20; // generous tensor dims (no OOB clipping wanted)
  g1[1] = (big & 0xFFFF) << 16;                        // tensor_dim0 lo16
  g1[2] = ((big >> 16) & 0xFFFF) | ((big & 0xFFFF) << 16); // dim0 hi | dim1 lo
  g1[3] = ((big >> 16) & 0xFFFF) | (32 << 16);         // dim1 hi | tile_dim0=32
  g1[4] = 64;                                          // tile_dim1=64, tile_dim2=0
  g1[5] = k_elems;                                     // tensor_dim0_stride lo32
  g1[6] = 0;                                           // stride0 hi | stride1 lo
  g1[7] = 0;
  i32x4 gz = {0, 0, 0, 0};
  i32x8 z8 = {0, 0, 0, 0, 0, 0, 0, 0};
  __builtin_amdgcn_tensor_load_to_lds(g0, g1, gz, gz, z8, 0);
}
#endif

// ---------------------------------------------------------------------------
// Tiled transpose + cast: in[K,N] f32 -> out[N,K] bf16 (batched over grid z).
// ---------------------------------------------------------------------------
__global__ __launch_bounds__(256)
void transpose_cast_kernel(const float* __restrict__ in, bf16_t* __restrict__ outp,
                           int K, int N) {
  __shared__ float tile[32][33];
  const size_t zo = (size_t)blockIdx.z * K * N;
  const float* inz = in + zo;
  bf16_t* outz = outp + zo;
  const int k0 = blockIdx.y * 32, n0 = blockIdx.x * 32;
  const int tx = threadIdx.x, ty = threadIdx.y;   // 32 x 8
  #pragma unroll
  for (int r = 0; r < 32; r += 8)
    tile[ty + r][tx] = inz[(size_t)(k0 + ty + r) * N + n0 + tx];
  __syncthreads();
  #pragma unroll
  for (int r = 0; r < 32; r += 8)
    outz[(size_t)(n0 + ty + r) * K + k0 + tx] = (bf16_t)tile[tx][ty + r];
}

// ---------------------------------------------------------------------------
// Fused LayerNorm (D=1024) + bf16 cast. One row per 256-thread block.
// ---------------------------------------------------------------------------
__global__ __launch_bounds__(256)
void ln_cast_kernel(const float* __restrict__ x, const float* __restrict__ g,
                    const float* __restrict__ bln, bf16_t* __restrict__ outp) {
  const int row = blockIdx.x, t = threadIdx.x;
  const float* xr = x + (size_t)row * DD;
  float4 v = *(const float4*)(xr + t * 4);
  float s  = v.x + v.y + v.z + v.w;
  float ss = v.x * v.x + v.y * v.y + v.z * v.z + v.w * v.w;
  #pragma unroll
  for (int off = 1; off < 32; off <<= 1) {
    s  += __shfl_xor(s,  off, 32);
    ss += __shfl_xor(ss, off, 32);
  }
  __shared__ float sb[8], ssb[8];
  if ((t & 31) == 0) { sb[t >> 5] = s; ssb[t >> 5] = ss; }
  __syncthreads();
  float tot = 0.f, tots = 0.f;
  #pragma unroll
  for (int i = 0; i < 8; i++) { tot += sb[i]; tots += ssb[i]; }
  const float mu  = tot * (1.0f / (float)DD);
  const float var = tots * (1.0f / (float)DD) - mu * mu;
  const float rs  = rsqrtf(var + 1e-5f);
  const int c = t * 4;
  float4 gv = *(const float4*)(g + c);
  float4 bv = *(const float4*)(bln + c);
  v4bf o;
  o[0] = (bf16_t)((v.x - mu) * rs * gv.x + bv.x);
  o[1] = (bf16_t)((v.y - mu) * rs * gv.y + bv.y);
  o[2] = (bf16_t)((v.z - mu) * rs * gv.z + bv.z);
  o[3] = (bf16_t)((v.w - mu) * rs * gv.w + bv.w);
  *(v4bf*)(outp + (size_t)row * DD + c) = o;
}

// ---------------------------------------------------------------------------
// bf16 WMMA GEMM with transposed weights:
//   out = act(A[M,K] * BT[N,K]^T + bias [+ resid])
// Block: 256 threads = 8 waves; block tile 128(M) x 64(N); wave tile 16x64
// (4 accumulators reusing one A fragment).  K-step 32.  B tile (64 rows x
// 32 k, 80B LDS pitch) double-buffered via TDM (or load/ds_store fallback).
// ---------------------------------------------------------------------------
__global__ __launch_bounds__(256)
void gemm_bf16_wmma(const bf16_t* __restrict__ A, const bf16_t* __restrict__ BT,
                    const float* __restrict__ bias, const float* __restrict__ resid,
                    float* __restrict__ outF, bf16_t* __restrict__ outB,
                    int M, int N, int K, int relu) {
  __shared__ bf16_t Bt[2][64][40];   // [buf][n][k] , 80B row pitch

  const int tid  = threadIdx.x;
  const int lane = tid & 31, wid = tid >> 5;
  const int lo = lane & 15, hi = lane >> 4;
  const int m0 = blockIdx.y * 128;
  const int n0 = blockIdx.x * 64;

  const bf16_t* bbase = BT + (size_t)n0 * K;            // 64 rows of K elems
  const bf16_t* aptr  = A + (size_t)(m0 + wid * 16 + lo) * K + hi * 8;

#if !USE_TDM
  const int srow = tid >> 2;            // 64 rows, 4 threads per row
  const int skof = (tid & 3) * 8;       // 8 bf16 (16B) per thread
#endif

  v8f acc[4];
  #pragma unroll
  for (int nn = 0; nn < 4; nn++) acc[nn] = (v8f){0, 0, 0, 0, 0, 0, 0, 0};

  const int nsteps = K / 32;
#if USE_TDM
  const unsigned lds0 = (unsigned)(size_t)(void*)&Bt[0][0][0];
  if (wid == 0)
    tdm_load_tile(lds0, (unsigned long long)(size_t)(const void*)bbase, K);
#else
  *(v8bf*)&Bt[0][srow][skof] = *(const v8bf*)(bbase + (size_t)srow * K + skof);
#endif

  for (int s = 0; s < nsteps; ++s) {
    const int cur = s & 1;
#if USE_TDM
    if (wid == 0) {
      if (s + 1 < nsteps) {
        tdm_load_tile(lds0 + (1 - cur) * (unsigned)sizeof(Bt[0]),
                      (unsigned long long)(size_t)(const void*)(bbase + (s + 1) * 32),
                      K);
        __builtin_amdgcn_s_wait_tensorcnt(1);  // tile s complete, s+1 in flight
      } else {
        __builtin_amdgcn_s_wait_tensorcnt(0);  // last tile complete
      }
    }
#else
    if (s + 1 < nsteps) {
      const bf16_t* bsrc = bbase + (s + 1) * 32;
      *(v8bf*)&Bt[1 - cur][srow][skof] = *(const v8bf*)(bsrc + (size_t)srow * K + skof);
    }
#endif
    __syncthreads();

    if (s + 4 < nsteps) __builtin_prefetch(aptr + (s + 4) * 32, 0, 3);
    ABfrag af;
    af.h[0] = *(const v8bf*)(aptr + s * 32);
    af.h[1] = *(const v8bf*)(aptr + s * 32 + 16);
    #pragma unroll
    for (int nn = 0; nn < 4; nn++) {
      ABfrag bfg;
      bfg.h[0] = *(const v8bf*)(&Bt[cur][nn * 16 + lo][hi * 8]);
      bfg.h[1] = *(const v8bf*)(&Bt[cur][nn * 16 + lo][hi * 8 + 16]);
      acc[nn] = wmma_bf16(af.v, bfg.v, acc[nn]);
    }
    __syncthreads();
  }

  #pragma unroll
  for (int nn = 0; nn < 4; nn++) {
    const int ocol = n0 + nn * 16 + lo;
    const float bb = bias[ocol];
    #pragma unroll
    for (int j = 0; j < 8; j++) {
      const int orow = m0 + wid * 16 + j + 8 * hi;
      float v = acc[nn][j] + bb;
      if (resid) v += resid[(size_t)orow * N + ocol];
      if (relu)  v = fmaxf(v, 0.f);
      if (outF)  outF[(size_t)orow * N + ocol] = v;
      if (outB)  outB[(size_t)orow * N + ocol] = (bf16_t)v;
    }
  }
}

// ---------------------------------------------------------------------------
// Causal flash attention. One wave per (b, h, 16-query tile).
// Q/K/V/Ctx in [B*T, H*64] bf16 layout (row stride D, head col offset h*64).
// ---------------------------------------------------------------------------
__global__ __launch_bounds__(32)
void attn_fa_kernel(const bf16_t* __restrict__ Q, const bf16_t* __restrict__ K,
                    const bf16_t* __restrict__ V, bf16_t* __restrict__ Ctx) {
  __shared__ bf16_t Pt[16][40];  // P tile [q][key]  (A-layout source)
  __shared__ bf16_t Vt[64][40];  // V tile transposed [d][key] (B-layout source)

  const int lane = threadIdx.x;
  const int lo = lane & 15, hi = lane >> 4;
  const int bid = blockIdx.x;
  const int qt = bid & 63;
  const int h  = (bid >> 6) & 15;
  const int b  = bid >> 10;
  const int q0 = qt << 4;
  const size_t base = (size_t)b * TT * DD + (size_t)h * HDM;

  ABfrag aq0, aq1;
  {
    const bf16_t* qp = Q + base + (size_t)(q0 + lo) * DD;
    aq0.h[0] = *(const v8bf*)(qp + hi * 8);
    aq0.h[1] = *(const v8bf*)(qp + hi * 8 + 16);
    aq1.h[0] = *(const v8bf*)(qp + 32 + hi * 8);
    aq1.h[1] = *(const v8bf*)(qp + 32 + hi * 8 + 16);
  }

  const v8f vzero = {0, 0, 0, 0, 0, 0, 0, 0};
  v8f cacc[4];
  #pragma unroll
  for (int nn = 0; nn < 4; nn++) cacc[nn] = vzero;
  float rm[8], rl[8];
  #pragma unroll
  for (int j = 0; j < 8; j++) { rm[j] = -1e30f; rl[j] = 0.f; }

  const int nkt = (qt >> 1) + 1;
  for (int kt = 0; kt < nkt; ++kt) {
    const int s0 = kt << 5;

    v8f sa[2];
    #pragma unroll
    for (int half = 0; half < 2; ++half) {
      const bf16_t* kp = K + base + (size_t)(s0 + half * 16 + lo) * DD;
      ABfrag bk0, bk1;
      bk0.h[0] = *(const v8bf*)(kp + hi * 8);
      bk0.h[1] = *(const v8bf*)(kp + hi * 8 + 16);
      bk1.h[0] = *(const v8bf*)(kp + 32 + hi * 8);
      bk1.h[1] = *(const v8bf*)(kp + 32 + hi * 8 + 16);
      v8f s = vzero;
      s = wmma_bf16(aq0.v, bk0.v, s);
      s = wmma_bf16(aq1.v, bk1.v, s);
      sa[half] = s;
    }

    __syncthreads();

    #pragma unroll
    for (int j = 0; j < 8; j++) {
      const int qi = q0 + j + 8 * hi;
      float v0 = sa[0][j] * 0.125f;   // 1/sqrt(64)
      float v1 = sa[1][j] * 0.125f;
      if (s0 + lo      > qi) v0 = -1e30f;
      if (s0 + 16 + lo > qi) v1 = -1e30f;
      float mx = fmaxf(v0, v1);
      #pragma unroll
      for (int off = 1; off < 16; off <<= 1) mx = fmaxf(mx, __shfl_xor(mx, off, 32));
      const float mnew = fmaxf(rm[j], mx);
      const float corr = __expf(rm[j] - mnew);
      rm[j] = mnew;
      const float p0 = __expf(v0 - mnew);
      const float p1 = __expf(v1 - mnew);
      float ps = p0 + p1;
      #pragma unroll
      for (int off = 1; off < 16; off <<= 1) ps += __shfl_xor(ps, off, 32);
      rl[j] = rl[j] * corr + ps;
      #pragma unroll
      for (int nn = 0; nn < 4; nn++) cacc[nn][j] *= corr;
      Pt[j + 8 * hi][lo]      = (bf16_t)p0;
      Pt[j + 8 * hi][16 + lo] = (bf16_t)p1;
    }

    {
      const bf16_t* vp = V + base + (size_t)(s0 + lane) * DD;
      #pragma unroll
      for (int c = 0; c < 8; c++) {
        v8bf vv = *(const v8bf*)(vp + c * 8);
        #pragma unroll
        for (int j = 0; j < 8; j++) Vt[c * 8 + j][lane] = vv[j];
      }
    }
    __syncthreads();

    ABfrag ap;
    ap.h[0] = *(const v8bf*)(&Pt[lo][hi * 8]);
    ap.h[1] = *(const v8bf*)(&Pt[lo][hi * 8 + 16]);
    #pragma unroll
    for (int nn = 0; nn < 4; nn++) {
      ABfrag bvf;
      bvf.h[0] = *(const v8bf*)(&Vt[nn * 16 + lo][hi * 8]);
      bvf.h[1] = *(const v8bf*)(&Vt[nn * 16 + lo][hi * 8 + 16]);
      cacc[nn] = wmma_bf16(ap.v, bvf.v, cacc[nn]);
    }
  }

  #pragma unroll
  for (int j = 0; j < 8; j++) {
    const float inv = 1.0f / rl[j];
    const size_t orow = base + (size_t)(q0 + j + 8 * hi) * DD;
    #pragma unroll
    for (int nn = 0; nn < 4; nn++)
      Ctx[orow + nn * 16 + lo] = (bf16_t)(cacc[nn][j] * inv);
  }
}

// ---------------------------------------------------------------------------
extern "C" void kernel_launch(void* const* d_in, const int* in_sizes, int n_in,
                              void* d_out, int out_size, void* d_ws, size_t ws_size,
                              hipStream_t stream) {
  (void)in_sizes; (void)n_in; (void)out_size; (void)ws_size;

  const float* x   = (const float*)d_in[0];
  const float* Wq  = (const float*)d_in[1];
  const float* bq  = (const float*)d_in[2];
  const float* Wk  = (const float*)d_in[3];
  const float* bk  = (const float*)d_in[4];
  const float* Wv  = (const float*)d_in[5];
  const float* bvb = (const float*)d_in[6];
  const float* Wo  = (const float*)d_in[7];
  const float* bo  = (const float*)d_in[8];
  const float* g1  = (const float*)d_in[9];
  const float* be1 = (const float*)d_in[10];
  const float* g2  = (const float*)d_in[11];
  const float* be2 = (const float*)d_in[12];
  const float* W1  = (const float*)d_in[13];
  const float* bf1 = (const float*)d_in[14];
  const float* W2  = (const float*)d_in[15];
  const float* bf2 = (const float*)d_in[16];
  float* out = (float*)d_out;

  char* ws = (char*)d_ws;
  const size_t MiB = (size_t)1 << 20;
  bf16_t* hbf  = (bf16_t*)(ws + 0 * MiB);    // 16 MiB  LN1 out (reused for LN2)
  bf16_t* wqT  = (bf16_t*)(ws + 16 * MiB);   // 2 MiB   [H*64, D] bf16
  bf16_t* wkT  = (bf16_t*)(ws + 18 * MiB);   // 2 MiB
  bf16_t* wvT  = (bf16_t*)(ws + 20 * MiB);   // 2 MiB
  bf16_t* woT  = (bf16_t*)(ws + 22 * MiB);   // 2 MiB   [D, D]
  bf16_t* w1T  = (bf16_t*)(ws + 24 * MiB);   // 8 MiB   [4D, D]
  bf16_t* w2T  = (bf16_t*)(ws + 32 * MiB);   // 8 MiB   [D, 4D]
  bf16_t* Qb   = (bf16_t*)(ws + 40 * MiB);   // 16 MiB
  bf16_t* Kb   = (bf16_t*)(ws + 56 * MiB);   // 16 MiB
  bf16_t* Vb   = (bf16_t*)(ws + 72 * MiB);   // 16 MiB
  bf16_t* ctxb = (bf16_t*)(ws + 88 * MiB);   // 16 MiB
  float*  xatt = (float*)(ws + 104 * MiB);   // 32 MiB
  bf16_t* actb = (bf16_t*)(ws + 40 * MiB);   // 64 MiB, overlays Q/K/V/ctx (dead by then)
  // total workspace footprint: 136 MiB

  dim3 t32x8(32, 8);
  // QKV: per-head [D,64] -> [64,D] ; flat result is [H*64, D] (concat-head rows)
  transpose_cast_kernel<<<dim3(2, 32, 16), t32x8, 0, stream>>>(Wq, wqT, DD, HDM);
  transpose_cast_kernel<<<dim3(2, 32, 16), t32x8, 0, stream>>>(Wk, wkT, DD, HDM);
  transpose_cast_kernel<<<dim3(2, 32, 16), t32x8, 0, stream>>>(Wv, wvT, DD, HDM);
  transpose_cast_kernel<<<dim3(32, 32, 1),  t32x8, 0, stream>>>(Wo, woT, DD, DD);
  transpose_cast_kernel<<<dim3(128, 32, 1), t32x8, 0, stream>>>(W1, w1T, DD, DFF);
  transpose_cast_kernel<<<dim3(32, 128, 1), t32x8, 0, stream>>>(W2, w2T, DFF, DD);

  // LN1
  ln_cast_kernel<<<MROWS, 256, 0, stream>>>(x, g1, be1, hbf);

  dim3 blk(256);
  dim3 gD(DD / 64, MROWS / 128);     // N=1024 GEMMs
  dim3 gF(DFF / 64, MROWS / 128);    // N=4096 GEMM

  // QKV projections -> bf16 [B*T, H*64]
  gemm_bf16_wmma<<<gD, blk, 0, stream>>>(hbf, wqT, bq, nullptr, nullptr, Qb,
                                         MROWS, DD, DD, 0);
  gemm_bf16_wmma<<<gD, blk, 0, stream>>>(hbf, wkT, bk, nullptr, nullptr, Kb,
                                         MROWS, DD, DD, 0);
  gemm_bf16_wmma<<<gD, blk, 0, stream>>>(hbf, wvT, bvb, nullptr, nullptr, Vb,
                                         MROWS, DD, DD, 0);

  // causal flash attention
  attn_fa_kernel<<<BB * HH * (TT / 16), 32, 0, stream>>>(Qb, Kb, Vb, ctxb);

  // out-proj + residual: xatt = x + ctx @ Wo + bo
  gemm_bf16_wmma<<<gD, blk, 0, stream>>>(ctxb, woT, bo, x, xatt, nullptr,
                                         MROWS, DD, DD, 0);

  // LN2
  ln_cast_kernel<<<MROWS, 256, 0, stream>>>(xatt, g2, be2, hbf);

  // FFN1: relu(h2 @ W1 + b1) -> bf16
  gemm_bf16_wmma<<<gF, blk, 0, stream>>>(hbf, w1T, bf1, nullptr, nullptr, actb,
                                         MROWS, DFF, DD, 1);

  // FFN2 + residual: out = xatt + act @ W2 + b2
  gemm_bf16_wmma<<<gD, blk, 0, stream>>>(actb, w2T, bf2, xatt, out, nullptr,
                                         MROWS, DD, DFF, 0);
}
